// MolNet_558345748859
// MI455X (gfx1250) — compile-verified, matrix operand order checked by reading, not run on hardware
//
#include <hip/hip_runtime.h>
#include <hip/hip_bf16.h>
#include <math.h>

#define D_MODEL 128
#define D_HEAD  64
#define N_HEADS 2
#define S_LEN   256
#define BATCH   256
#define D_FF    512
#define N_TRAJ  20000
#define USERS   200
#define TOKENS  (BATCH * S_LEN)

typedef __attribute__((ext_vector_type(16))) __bf16 v16bf;
typedef __attribute__((ext_vector_type(8)))  float  v8f;

// ---------------------------------------------------------------- helpers
__device__ __forceinline__ unsigned short f2bf(float f) {
  union { float f; unsigned u; } c; c.f = f;
  unsigned u = c.u + 0x7FFFu + ((c.u >> 16) & 1u);   // round-to-nearest-even
  return (unsigned short)(u >> 16);
}
__device__ __forceinline__ float bf2f(unsigned short h) {
  union { unsigned u; float f; } c; c.u = ((unsigned)h) << 16; return c.f;
}

// Load one 16x32 bf16 WMMA fragment from an LDS tile laid out [row][k],
// row-major with `stride` ushorts per row. Works for A ([M][K]) and for
// B when staged transposed as [N][K].
// Lane map (ISA 7.12.2, 16-bit A 16x32): lanes 0-15 -> rows, k-half 0;
// lanes 16-31 -> same rows, k-half 1. Elements 0..7 = k contiguous,
// elements 8..15 = k+16 contiguous  ->  two 16-byte ds loads.
__device__ __forceinline__ v16bf load_frag(const unsigned short* base,
                                           int row0, int stride, int k0) {
  const int lane = threadIdx.x & 31;
  const int r  = row0 + (lane & 15);
  const int kb = k0 + ((lane >> 4) << 3);
  union { v16bf v; uint4 q[2]; } f;
  f.q[0] = *(const uint4*)(base + r * stride + kb);
  f.q[1] = *(const uint4*)(base + r * stride + kb + 16);
  return f.v;
}

#define WMMA_BF16(a, b, c) \
  __builtin_amdgcn_wmma_f32_16x16x32_bf16(false, (a), false, (b), (short)0, (c), false, false)

// ---------------------------------------------------------------- 1) embed + fusion + posenc
__global__ void k_embed_fuse(const float* __restrict__ grid_emb,
                             const int*   __restrict__ input_seq,
                             const int*   __restrict__ time_seq,
                             const int*   __restrict__ state_seq,
                             const float* __restrict__ time_table,
                             const float* __restrict__ state_table,
                             const float* __restrict__ W_fc,
                             const float* __restrict__ b_fc,
                             float* __restrict__ x) {
  __shared__ float a[192];
  const int tok = blockIdx.x;
  const int s   = tok & (S_LEN - 1);
  const int tid = threadIdx.x;
  const int g   = input_seq[tok];
  if (tid < 128)      a[tid] = (g >= 0) ? grid_emb[(size_t)g * D_MODEL + tid] : 0.f;
  else if (tid < 160) a[tid] = time_table[time_seq[tok] * 32 + (tid - 128)];
  else if (tid < 192) a[tid] = state_table[state_seq[tok] * 32 + (tid - 160)];
  __syncthreads();
  if (tid < 128) {
    float acc = b_fc[tid];
#pragma unroll 8
    for (int j = 0; j < 192; ++j) acc += a[j] * W_fc[j * D_MODEL + tid];
    float y = tanhf(acc);
    int p = tid >> 1;
    float ang = (float)s * __powf(10000.f, -(float)(2 * p) / 128.f);
    y += (tid & 1) ? __cosf(ang) : __sinf(ang);
    x[(size_t)tok * D_MODEL + tid] = y;
  }
}

// ---------------------------------------------------------------- 2) generic bf16 WMMA GEMM
// C[M,N] = A[M,K] @ B[K,N].  64x128 block tile, K-step 32. 8 waves: 4x2 grid,
// each wave 16x64 (4 WMMA n-tiles).
__global__ void k_gemm(const float* __restrict__ A, const float* __restrict__ B,
                       float* __restrict__ C, int N, int K) {
  __shared__ __align__(16) unsigned short sA[64 * 32];
  __shared__ __align__(16) unsigned short sBt[128 * 32];
  const int tid = threadIdx.x;
  const int m0  = blockIdx.x * 64;
  const int n0  = blockIdx.y * 128;
  const int wid = tid >> 5, lane = tid & 31;
  const int lh = lane >> 4, lnn = lane & 15;
  const int wm = (wid >> 1) * 16;
  const int wn = (wid & 1) * 64;
  v8f acc[4] = {};
  for (int k0 = 0; k0 < K; k0 += 32) {
    __syncthreads();
    if (k0 + 32 < K) __builtin_prefetch(&A[(size_t)m0 * K + k0 + 32], 0, 1);
    for (int i = tid; i < 64 * 32; i += 256) {
      int m = i >> 5, kk = i & 31;
      sA[m * 32 + kk] = f2bf(A[(size_t)(m0 + m) * K + k0 + kk]);
    }
    for (int i = tid; i < 32 * 128; i += 256) {
      int kk = i >> 7, n = i & 127;
      sBt[n * 32 + kk] = f2bf(B[(size_t)(k0 + kk) * N + n0 + n]);
    }
    __syncthreads();
    v16bf af = load_frag(sA, wm, 32, 0);
#pragma unroll
    for (int t = 0; t < 4; ++t) {
      v16bf bf = load_frag(sBt, wn + t * 16, 32, 0);
      acc[t] = WMMA_BF16(af, bf, acc[t]);
    }
  }
  const int mr = m0 + wm + lh * 8;
  const int nc = n0 + wn + lnn;
#pragma unroll
  for (int t = 0; t < 4; ++t)
#pragma unroll
    for (int i = 0; i < 8; ++i)
      C[(size_t)(mr + i) * N + nc + t * 16] = acc[t][i];
}

// ---------------------------------------------------------------- 3) GEMM + residual + LayerNorm
// out = LN(A @ B + res).  N fixed at 128 (full row per block -> LN is block-local).
__global__ void k_gemm_ln(const float* __restrict__ A, const float* __restrict__ B,
                          const float* __restrict__ res,
                          const float* __restrict__ gamma, const float* __restrict__ beta,
                          float* __restrict__ out, int K) {
  __shared__ __align__(16) unsigned short sA[64 * 32];
  __shared__ __align__(16) unsigned short sBt[128 * 32];
  __shared__ float yT[64 * 128];
  const int tid = threadIdx.x;
  const int m0  = blockIdx.x * 64;
  const int wid = tid >> 5, lane = tid & 31;
  const int lh = lane >> 4, lnn = lane & 15;
  const int wm = (wid >> 1) * 16;
  const int wn = (wid & 1) * 64;
  v8f acc[4] = {};
  for (int k0 = 0; k0 < K; k0 += 32) {
    __syncthreads();
    for (int i = tid; i < 64 * 32; i += 256) {
      int m = i >> 5, kk = i & 31;
      sA[m * 32 + kk] = f2bf(A[(size_t)(m0 + m) * K + k0 + kk]);
    }
    for (int i = tid; i < 32 * 128; i += 256) {
      int kk = i >> 7, n = i & 127;
      sBt[n * 32 + kk] = f2bf(B[(size_t)(k0 + kk) * 128 + n]);
    }
    __syncthreads();
    v16bf af = load_frag(sA, wm, 32, 0);
#pragma unroll
    for (int t = 0; t < 4; ++t) {
      v16bf bf = load_frag(sBt, wn + t * 16, 32, 0);
      acc[t] = WMMA_BF16(af, bf, acc[t]);
    }
  }
  __syncthreads();
#pragma unroll
  for (int t = 0; t < 4; ++t)
#pragma unroll
    for (int i = 0; i < 8; ++i) {
      int ml = wm + lh * 8 + i;
      int n  = wn + t * 16 + lnn;
      yT[ml * 128 + n] = acc[t][i] + res[(size_t)(m0 + ml) * 128 + n];
    }
  __syncthreads();
  if (tid < 64) {
    float mean = 0.f, sq = 0.f;
    for (int c = 0; c < 128; ++c) { float v = yT[tid * 128 + c]; mean += v; sq += v * v; }
    mean *= (1.f / 128.f);
    float var = sq * (1.f / 128.f) - mean * mean;
    float inv = rsqrtf(var + 1e-5f);
    for (int c = 0; c < 128; ++c)
      out[(size_t)(m0 + tid) * 128 + c] = (yT[tid * 128 + c] - mean) * inv * gamma[c] + beta[c];
  }
}

// ---------------------------------------------------------------- 4) fused attention
// One block per (batch, head, 64-row q-tile). Scores live in LDS (bf16), never in HBM.
__global__ void k_attn(const float* __restrict__ qb, const float* __restrict__ kbuf,
                       const float* __restrict__ vbuf, const int* __restrict__ input_seq,
                       float* __restrict__ ctx) {
  __shared__ __align__(16) unsigned short sQ[64 * 64];
  __shared__ __align__(16) unsigned short sKV[64 * 64];
  __shared__ __align__(16) unsigned short sP[64 * 256];
  __shared__ float maskb[256];
  const int tid = threadIdx.x;
  const int idx = blockIdx.x;
  const int b = idx >> 3, h = (idx >> 2) & 1, qt = idx & 3;
  const int wid = tid >> 5, lane = tid & 31;
  const int lh = lane >> 4, lnn = lane & 15;
  const size_t rowbase = (size_t)(b * S_LEN + qt * 64);
  const int wm = (wid >> 1) * 16;   // q-row offset within tile
  const int wn = (wid & 1) * 32;    // key/d offset within tile

  for (int i = tid; i < 64 * 64; i += 256) {
    int r = i >> 6, c = i & 63;
    sQ[i] = f2bf(qb[(rowbase + r) * D_MODEL + h * D_HEAD + c]);
  }
  maskb[tid] = (input_seq[b * S_LEN + tid] >= 0) ? 0.f : -1e9f;

  // ---- scores: S = Q K^T * scale + mask, bf16 into sP
  for (int kt = 0; kt < 4; ++kt) {
    __syncthreads();
    for (int i = tid; i < 64 * 64; i += 256) {
      int r = i >> 6, c = i & 63;   // sKV[key][d] == transposed-B layout for QK^T
      sKV[i] = f2bf(kbuf[((size_t)(b * S_LEN + kt * 64 + r)) * D_MODEL + h * D_HEAD + c]);
    }
    __syncthreads();
    v8f acc[2] = {};
    for (int ks = 0; ks < 64; ks += 32) {
      v16bf af = load_frag(sQ, wm, 64, ks);
#pragma unroll
      for (int t = 0; t < 2; ++t) {
        v16bf bf = load_frag(sKV, wn + t * 16, 64, ks);
        acc[t] = WMMA_BF16(af, bf, acc[t]);
      }
    }
#pragma unroll
    for (int t = 0; t < 2; ++t)
#pragma unroll
      for (int i = 0; i < 8; ++i) {
        int m   = wm + lh * 8 + i;
        int key = kt * 64 + wn + t * 16 + lnn;
        sP[m * 256 + key] = f2bf(acc[t][i] * 0.125f + maskb[key]);
      }
  }
  __syncthreads();

  // ---- softmax per q row (64 rows, serial over 256 keys)
  if (tid < 64) {
    float mx = -3e38f;
    for (int j = 0; j < 256; ++j) mx = fmaxf(mx, bf2f(sP[tid * 256 + j]));
    float sum = 0.f;
    for (int j = 0; j < 256; ++j) {
      float e = __expf(bf2f(sP[tid * 256 + j]) - mx);
      sum += e;
      sP[tid * 256 + j] = f2bf(e);
    }
    float inv = 1.f / sum;
    for (int j = 0; j < 256; ++j)
      sP[tid * 256 + j] = f2bf(bf2f(sP[tid * 256 + j]) * inv);
  }

  // ---- O = P V  (V staged transposed: sKV[d][key])
  v8f oacc[2] = {};
  for (int kt = 0; kt < 4; ++kt) {
    __syncthreads();
    for (int i = tid; i < 64 * 64; i += 256) {
      int d = i >> 6, kl = i & 63;
      sKV[d * 64 + kl] = f2bf(vbuf[((size_t)(b * S_LEN + kt * 64 + kl)) * D_MODEL + h * D_HEAD + d]);
    }
    __syncthreads();
    for (int ks = 0; ks < 64; ks += 32) {
      v16bf af = load_frag(sP, wm, 256, kt * 64 + ks);
#pragma unroll
      for (int t = 0; t < 2; ++t) {
        v16bf bf = load_frag(sKV, wn + t * 16, 64, ks);
        oacc[t] = WMMA_BF16(af, bf, oacc[t]);
      }
    }
  }
#pragma unroll
  for (int t = 0; t < 2; ++t)
#pragma unroll
    for (int i = 0; i < 8; ++i)
      ctx[(rowbase + wm + lh * 8 + i) * D_MODEL + h * D_HEAD + wn + t * 16 + lnn] = oacc[t][i];
}

// ---------------------------------------------------------------- 5) fully fused FFN + LN
// x = LN(relu(x W1 + b1) W2 + b2 + x). 32-row tile; 512-wide hidden lives in LDS.
__global__ void k_ffn_ln(const float* __restrict__ xin, float* __restrict__ xout,
                         const float* __restrict__ W1, const float* __restrict__ b1,
                         const float* __restrict__ W2, const float* __restrict__ b2,
                         const float* __restrict__ gamma, const float* __restrict__ beta) {
  __shared__ __align__(16) unsigned short smem[28672];   // 56 KB arena
  unsigned short* sB1t = smem;                 // [512][32] (phase 1)
  unsigned short* sH   = smem;                 // [32][512] hidden bf16 (phase 2, reuses sB1t)
  unsigned short* sA   = smem + 16384;         // [32][32]  (phase 1)
  unsigned short* sB2t = smem + 16384;         // [128][32] (phase 2, reuses sA region)
  float* yT = (float*)(smem + 20480);          // [32][128] fp32 (phase 2 epilogue)

  const int tid = threadIdx.x, wid = tid >> 5, lane = tid & 31;
  const int lh = lane >> 4, lnn = lane & 15;
  const int m0 = blockIdx.x * 32;

  // ---- phase 1: hidden = relu(x @ W1 + b1), 32x512, waves 2x4 (16x128 each)
  const int wm1 = (wid >> 2) * 16;
  const int wn1 = (wid & 3) * 128;
  v8f acc1[8] = {};
  for (int k0 = 0; k0 < 128; k0 += 32) {
    __syncthreads();
    for (int i = tid; i < 32 * 32; i += 256) {
      int m = i >> 5, kk = i & 31;
      sA[m * 32 + kk] = f2bf(xin[(size_t)(m0 + m) * 128 + k0 + kk]);
    }
    for (int i = tid; i < 16384; i += 256) {
      int kk = i >> 9, n = i & 511;
      sB1t[n * 32 + kk] = f2bf(W1[(size_t)(k0 + kk) * 512 + n]);
    }
    __syncthreads();
    v16bf af = load_frag(sA, wm1, 32, 0);
#pragma unroll
    for (int t = 0; t < 8; ++t) {
      v16bf bf = load_frag(sB1t, wn1 + t * 16, 32, 0);
      acc1[t] = WMMA_BF16(af, bf, acc1[t]);
    }
  }
  __syncthreads();                       // last reads of sB1t done -> safe to write sH
#pragma unroll
  for (int t = 0; t < 8; ++t)
#pragma unroll
    for (int i = 0; i < 8; ++i) {
      int m = wm1 + lh * 8 + i;
      int n = wn1 + t * 16 + lnn;
      float hv = acc1[t][i] + b1[n];
      sH[m * 512 + n] = f2bf(hv > 0.f ? hv : 0.f);
    }

  // ---- phase 2: y = hidden @ W2 + b2 + x, then LN. waves 2x4 (16x32 each)
  const int wm2 = (wid >> 2) * 16;
  const int wn2 = (wid & 3) * 32;
  v8f acc2[2] = {};
  for (int k0 = 0; k0 < 512; k0 += 32) {
    __syncthreads();
    for (int i = tid; i < 128 * 32; i += 256) {
      int kk = i >> 7, n = i & 127;
      sB2t[n * 32 + kk] = f2bf(W2[(size_t)(k0 + kk) * 128 + n]);
    }
    __syncthreads();
    v16bf af = load_frag(sH, wm2, 512, k0);
#pragma unroll
    for (int t = 0; t < 2; ++t) {
      v16bf bf = load_frag(sB2t, wn2 + t * 16, 32, 0);
      acc2[t] = WMMA_BF16(af, bf, acc2[t]);
    }
  }
  __syncthreads();
#pragma unroll
  for (int t = 0; t < 2; ++t)
#pragma unroll
    for (int i = 0; i < 8; ++i) {
      int m = wm2 + lh * 8 + i;
      int n = wn2 + t * 16 + lnn;
      yT[m * 128 + n] = acc2[t][i] + b2[n] + xin[(size_t)(m0 + m) * 128 + n];
    }
  __syncthreads();
  if (tid < 32) {
    float mean = 0.f, sq = 0.f;
    for (int c = 0; c < 128; ++c) { float v = yT[tid * 128 + c]; mean += v; sq += v * v; }
    mean *= (1.f / 128.f);
    float var = sq * (1.f / 128.f) - mean * mean;
    float inv = rsqrtf(var + 1e-5f);
    for (int c = 0; c < 128; ++c)
      xout[(size_t)(m0 + tid) * 128 + c] = (yT[tid * 128 + c] - mean) * inv * gamma[c] + beta[c];
  }
}

// ---------------------------------------------------------------- 6) mean pool over S
__global__ void k_pool(const float* __restrict__ x, float* __restrict__ pooled) {
  const int b = blockIdx.x, d = threadIdx.x;
  float s = 0.f;
  for (int t = 0; t < S_LEN; ++t) s += x[((size_t)b * S_LEN + t) * D_MODEL + d];
  pooled[b * D_MODEL + d] = s * (1.f / S_LEN);
}

// ---------------------------------------------------------------- 7) global elastic attention
__global__ void k_gattn(const float* __restrict__ traj, const int* __restrict__ input_index,
                        float* __restrict__ scores, float* __restrict__ gctx) {
  __shared__ float qv[128];
  __shared__ float red[256];
  const int b = blockIdx.x, tid = threadIdx.x;
  const int qi = input_index[b];
  if (tid < 128) qv[tid] = traj[(size_t)qi * 128 + tid];
  __syncthreads();
  float lmax = -3e38f;
  for (int j = tid; j < N_TRAJ; j += 256) {
    float s = 0.f;
    for (int d = 0; d < 128; ++d) s += qv[d] * traj[(size_t)j * 128 + d];
    scores[(size_t)b * N_TRAJ + j] = s;
    lmax = fmaxf(lmax, s);
  }
  red[tid] = lmax; __syncthreads();
  for (int off = 128; off; off >>= 1) {
    if (tid < off) red[tid] = fmaxf(red[tid], red[tid + off]);
    __syncthreads();
  }
  const float mx = red[0];
  __syncthreads();
  float lsum = 0.f;
  for (int j = tid; j < N_TRAJ; j += 256) {
    float e = __expf(scores[(size_t)b * N_TRAJ + j] - mx);
    scores[(size_t)b * N_TRAJ + j] = e;
    lsum += e;
  }
  red[tid] = lsum; __syncthreads();
  for (int off = 128; off; off >>= 1) {
    if (tid < off) red[tid] += red[tid + off];
    __syncthreads();
  }
  const float inv = 1.f / red[0];
  __syncthreads();
  if (tid < 128) {
    float c = 0.f;
    for (int j = 0; j < N_TRAJ; ++j) c += scores[(size_t)b * N_TRAJ + j] * traj[(size_t)j * 128 + tid];
    gctx[b * 128 + tid] = c * inv;
  }
}

// ---------------------------------------------------------------- 8) classifier + log-softmax
__global__ void k_cls(const float* __restrict__ pooled, const float* __restrict__ gctx,
                      const float* __restrict__ Wc1, const float* __restrict__ bc1,
                      const float* __restrict__ Wc2, const float* __restrict__ bc2,
                      float* __restrict__ out) {
  __shared__ float cat[256];
  __shared__ float hid[128];
  __shared__ float logit[USERS];
  __shared__ float stat[2];
  const int b = blockIdx.x, tid = threadIdx.x;
  cat[tid] = (tid < 128) ? pooled[b * 128 + tid] : gctx[b * 128 + tid - 128];
  __syncthreads();
  if (tid < 128) {
    float h = bc1[tid];
    for (int j = 0; j < 256; ++j) h += cat[j] * Wc1[j * 128 + tid];
    hid[tid] = h > 0.f ? h : 0.f;
  }
  __syncthreads();
  if (tid < USERS) {
    float z = bc2[tid];
    for (int j = 0; j < 128; ++j) z += hid[j] * Wc2[j * USERS + tid];
    logit[tid] = z;
  }
  __syncthreads();
  if (tid == 0) {
    float mx = -3e38f;
    for (int j = 0; j < USERS; ++j) mx = fmaxf(mx, logit[j]);
    float s = 0.f;
    for (int j = 0; j < USERS; ++j) s += __expf(logit[j] - mx);
    stat[0] = mx; stat[1] = __logf(s);
  }
  __syncthreads();
  if (tid < USERS) out[b * USERS + tid] = logit[tid] - stat[0] - stat[1];
}

// ---------------------------------------------------------------- launch
extern "C" void kernel_launch(void* const* d_in, const int* in_sizes, int n_in,
                              void* d_out, int out_size, void* d_ws, size_t ws_size,
                              hipStream_t stream) {
  const float* grid_emb    = (const float*)d_in[0];
  const float* traj_emb    = (const float*)d_in[1];
  const int*   input_seq   = (const int*)d_in[2];
  const int*   time_seq    = (const int*)d_in[3];
  const int*   state_seq   = (const int*)d_in[4];
  const int*   input_index = (const int*)d_in[5];
  const float* time_table  = (const float*)d_in[6];
  const float* state_table = (const float*)d_in[7];
  const float* W_fc = (const float*)d_in[8];
  const float* b_fc = (const float*)d_in[9];
  const float* Wq   = (const float*)d_in[10];
  const float* Wk   = (const float*)d_in[11];
  const float* Wv   = (const float*)d_in[12];
  const float* Wo   = (const float*)d_in[13];
  const float* ln1_g = (const float*)d_in[14];
  const float* ln1_b = (const float*)d_in[15];
  const float* W1   = (const float*)d_in[16];
  const float* b1   = (const float*)d_in[17];
  const float* W2   = (const float*)d_in[18];
  const float* b2   = (const float*)d_in[19];
  const float* ln2_g = (const float*)d_in[20];
  const float* ln2_b = (const float*)d_in[21];
  const float* Wc1  = (const float*)d_in[22];
  const float* bc1  = (const float*)d_in[23];
  const float* Wc2  = (const float*)d_in[24];
  const float* bc2  = (const float*)d_in[25];
  float* outp = (float*)d_out;

  const size_t NM = (size_t)TOKENS * D_MODEL;      // 8.39M floats per activation buffer
  float* x  = (float*)d_ws;
  float* qb = x + NM;
  float* kb = qb + NM;
  float* vb = kb + NM;
  float* cb = vb + NM;
  float* scores = qb;                  // reuse q buffer after the transformer layers
  float* pooled = kb;                  // reuse k buffer
  float* gctx   = kb + BATCH * D_MODEL;

  k_embed_fuse<<<TOKENS, 256, 0, stream>>>(grid_emb, input_seq, time_seq, state_seq,
                                           time_table, state_table, W_fc, b_fc, x);
  for (int l = 0; l < 2; ++l) {
    k_gemm<<<dim3(TOKENS / 64, 1), 256, 0, stream>>>(x, Wq + (size_t)l * 128 * 128, qb, 128, 128);
    k_gemm<<<dim3(TOKENS / 64, 1), 256, 0, stream>>>(x, Wk + (size_t)l * 128 * 128, kb, 128, 128);
    k_gemm<<<dim3(TOKENS / 64, 1), 256, 0, stream>>>(x, Wv + (size_t)l * 128 * 128, vb, 128, 128);
    k_attn<<<BATCH * N_HEADS * 4, 256, 0, stream>>>(qb, kb, vb, input_seq, cb);
    k_gemm_ln<<<TOKENS / 64, 256, 0, stream>>>(cb, Wo + (size_t)l * 128 * 128, x,
                                               ln1_g + l * 128, ln1_b + l * 128, x, 128);
    k_ffn_ln<<<TOKENS / 32, 256, 0, stream>>>(x, x, W1 + (size_t)l * 128 * 512, b1 + l * 512,
                                              W2 + (size_t)l * 512 * 128, b2 + l * 128,
                                              ln2_g + l * 128, ln2_b + l * 128);
  }
  k_pool<<<BATCH, 128, 0, stream>>>(x, pooled);
  k_gattn<<<BATCH, 256, 0, stream>>>(traj_emb, input_index, scores, gctx);
  k_cls<<<BATCH, 256, 0, stream>>>(pooled, gctx, Wc1, bc1, Wc2, bc2, outp);
}
